// Branch_62989990363328
// MI455X (gfx1250) — compile-verified, matrix operand order checked by reading, not run on hardware
//
#include <hip/hip_runtime.h>
#include <hip/hip_bf16.h>
#include <math.h>

typedef __attribute__((ext_vector_type(16))) _Float16 v16h;
typedef __attribute__((ext_vector_type(8)))  float    v8f;
typedef __attribute__((ext_vector_type(4)))  unsigned u32x4;
typedef __attribute__((ext_vector_type(8)))  int      i32x8;
typedef __attribute__((ext_vector_type(4)))  int      i32x4;

#define WMMA_F16(a, b, c) \
  __builtin_amdgcn_wmma_f32_16x16x32_f16(false, (a), false, (b), (short)0, (c), false, false)

#if defined(__has_builtin)
#if __has_builtin(__builtin_amdgcn_tensor_load_to_lds) && __has_builtin(__builtin_amdgcn_s_wait_tensorcnt)
#define HAVE_TDM 1
#endif
#endif
#ifndef HAVE_TDM
#define HAVE_TDM 0
#endif
#if __has_include(<hip/amd_detail/amd_gfx1250_TDM.h>)
#define TDM_SIX 1
#else
#define TDM_SIX 0
#endif

// K index inside a 16x32 f16 A/B fragment: lane group g, half index i.
// g=0: halves 0..7 -> K0..7, 8..15 -> K16..23 ; g=1: K8..15 / K24..31.
__device__ __forceinline__ int fragk(int i, int g) { return (i < 8 ? i : i + 8) + (g << 3); }

// Row-major operand fragment: halves 0..7 = p[g*8..], 8..15 = p[16+g*8..] (two b128 loads)
__device__ __forceinline__ v16h load_row16(const _Float16* __restrict__ p, int g) {
  v16h r;
#pragma unroll
  for (int i = 0; i < 8; ++i) { r[i] = p[(g << 3) + i]; r[i + 8] = p[16 + (g << 3) + i]; }
  return r;
}
// Fragment for K padded to 32 with rows 16..31 == 0; p points to a 16-element row.
__device__ __forceinline__ v16h load_t16(const _Float16* __restrict__ p, int g) {
  v16h r;
#pragma unroll
  for (int i = 0; i < 8; ++i) { r[i] = p[(g << 3) + i]; r[i + 8] = (_Float16)0.0f; }
  return r;
}

__device__ __forceinline__ float gelu_f(float x) {
  return 0.5f * x * (1.0f + erff(x * 0.70710678118654752f));
}
__device__ __forceinline__ float gelu_bwd_f(float x) {
  return 0.5f * (1.0f + erff(x * 0.70710678118654752f)) +
         x * __expf(-0.5f * x * x) * 0.3989422804014327f;
}
__device__ __forceinline__ float silu_f(float x) { return x / (1.0f + __expf(-x)); }

// ---------------------------------------------------------------------------
// TDM: 2D tile load Global->LDS (data_size=4B).  D# packed per CDNA5 ISA ch.8.
// ---------------------------------------------------------------------------
__device__ __forceinline__ void tdm_load_2d_f32(unsigned lds_off, const void* gaddr,
                                                unsigned tile_d0, unsigned tile_d1,
                                                unsigned tensor_d0, unsigned tensor_d1,
                                                unsigned stride_d0) {
#if HAVE_TDM
  unsigned long long ga = (unsigned long long)(size_t)gaddr;
  u32x4 g0;
  g0[0] = 1u;                                            // count=1 (valid user D#)
  g0[1] = lds_off;                                       // lds_addr
  g0[2] = (unsigned)(ga & 0xFFFFFFFFu);                  // global_addr[31:0]
  g0[3] = (unsigned)((ga >> 32) & 0x01FFFFFFu) | (2u << 30);  // addr[56:32] | type=2
  i32x8 g1;
  g1[0] = (int)(2u << 16);                               // wg_mask=0, data_size=4B
  g1[1] = (int)((tensor_d0 & 0xFFFFu) << 16);            // tensor_dim0[15:0]
  g1[2] = (int)(((tensor_d0 >> 16) & 0xFFFFu) | ((tensor_d1 & 0xFFFFu) << 16));
  g1[3] = (int)(((tensor_d1 >> 16) & 0xFFFFu) | ((tile_d0 & 0xFFFFu) << 16));
  g1[4] = (int)(tile_d1 & 0xFFFFu);                      // tile_dim1 (tile_dim2=0)
  g1[5] = (int)stride_d0;                                // tensor_dim0_stride[31:0]
  g1[6] = 0; g1[7] = 0;
  i32x4 z4; z4[0] = 0; z4[1] = 0; z4[2] = 0; z4[3] = 0;
#if TDM_SIX
  i32x8 z8; for (int i = 0; i < 8; ++i) z8[i] = 0;
  __builtin_amdgcn_tensor_load_to_lds(g0, g1, z4, z4, z8, 0);
#else
  __builtin_amdgcn_tensor_load_to_lds(g0, g1, z4, z4, 0);
#endif
#else
  (void)lds_off; (void)gaddr; (void)tile_d0; (void)tile_d1;
  (void)tensor_d0; (void)tensor_d1; (void)stride_d0;
#endif
}

// ---------------------------------------------------------------------------
// Elementwise converters / data reorg
// ---------------------------------------------------------------------------
__global__ void cvt_f32_to_f16(const float* __restrict__ in, _Float16* __restrict__ out, int n) {
  int i = blockIdx.x * 256 + threadIdx.x;
  if (i < n) out[i] = (_Float16)in[i];
}

// zero-padded f16 input: xpad[b][iy 0..29][ix 0..29][c], halo = 5
__global__ void pad_input(const float* __restrict__ x, _Float16* __restrict__ xpad) {
  int i = blockIdx.x * 256 + threadIdx.x;
  if (i >= 2 * 30 * 30 * 64) return;
  const int c = i & 63;
  const int ix = (i >> 6) % 30;
  const int iy = (i >> 6) / 30 % 30;
  const int b = i / (30 * 30 * 64);
  float v = 0.0f;
  if (iy >= 5 && iy < 25 && ix >= 5 && ix < 25)
    v = x[((size_t)b * 400 + (iy - 5) * 20 + (ix - 5)) * 64 + c];
  xpad[i] = (_Float16)v;
}

// conv weight [OUT,64,kk,kk] f32 -> tap-major f16 wt[tap][oc][ic]
__global__ void reorg_convw(const float* __restrict__ w, _Float16* __restrict__ out, int k2) {
  size_t i = (size_t)blockIdx.x * 256 + threadIdx.x;
  if (i >= (size_t)2048 * 64 * k2) return;
  const int tap = (int)(i % k2);
  const size_t oc_ic = i / k2;
  const int ic = (int)(oc_ic & 63);
  const int oc = (int)(oc_ic >> 6);
  out[((size_t)tap * 2048 + oc) * 64 + ic] = (_Float16)w[i];
}

// TTT state in f16, three layouts + f32 biases, broadcast over 16 (b,h)
__global__ void init_ttt_state(const float* __restrict__ W1, const float* __restrict__ b1,
                               const float* __restrict__ W2, const float* __restrict__ b2,
                               _Float16* __restrict__ W1c, _Float16* __restrict__ W2r,
                               _Float16* __restrict__ W2c,
                               float* __restrict__ b1s, float* __restrict__ b2s) {
  size_t i = (size_t)blockIdx.x * 256 + threadIdx.x;
  const size_t per = 1024 * 256;
  if (i < 16 * per) {
    const int bh = (int)(i / per);
    const size_t e = i % per;
    const int h = bh & 7;
    { // W1c[inner n][hd k] = W1[h][k][n]
      const int n = (int)(e / 256), kq = (int)(e % 256);
      W1c[i] = (_Float16)W1[((size_t)h * 256 + kq) * 1024 + n];
    }
    { // W2r[inner m][hd n] = W2[h][m][n]
      W2r[i] = (_Float16)W2[(size_t)h * per + e];
    }
    { // W2c[hd n][inner k] = W2[h][k][n]
      const int n = (int)(e / 1024), kq = (int)(e % 1024);
      W2c[i] = (_Float16)W2[((size_t)h * 1024 + kq) * 256 + n];
    }
  }
  if (i < 16 * 1024) { int bh = (int)(i / 1024); int e = (int)(i % 1024); b1s[i] = b1[(bh & 7) * 1024 + e]; }
  if (i < 16 * 256)  { int bh = (int)(i / 256);  int e = (int)(i % 256);  b2s[i] = b2[(bh & 7) * 256 + e]; }
}

// ---------------------------------------------------------------------------
// Conv ensemble (k=1..11, SAME) + mean + SiLU as implicit-GEMM WMMA.
// All operand loads are contiguous b128 f16 (padded input + tap-major weights).
// grid (50 pos-tiles, 32 oc-groups), block 128 (4 waves x 16 oc).
// ---------------------------------------------------------------------------
__global__ void conv_ensemble_silu(const _Float16* __restrict__ xpad,
                                   const _Float16* __restrict__ wt,
                                   const float* __restrict__ cb0, const float* __restrict__ cb1,
                                   const float* __restrict__ cb2, const float* __restrict__ cb3,
                                   const float* __restrict__ cb4, const float* __restrict__ cb5,
                                   _Float16* __restrict__ xb_h) {
  const int lane = threadIdx.x & 31, wave = threadIdx.x >> 5;
  const int lm = lane & 15, g = lane >> 4;
  const int p0 = blockIdx.x * 16;
  const int oc0 = blockIdx.y * 64 + wave * 16;

  const int pos = p0 + lm;
  const int bb = pos / 400;
  const int t = pos % 400;
  const int ii = t / 20, jj = t % 20;

  const _Float16* __restrict__ wrow = wt + (size_t)(oc0 + lm) * 64;  // + tap*2048*64
  v8f acc = {};
  int tap = 0;
  for (int kern = 0; kern < 6; ++kern) {
    const int kk = 2 * kern + 1, ctr = kern;
    for (int dy = 0; dy < kk; ++dy) {
      const int yy = ii + dy - ctr + 5;
      for (int dx = 0; dx < kk; ++dx) {
        const int xx = jj + dx - ctr + 5;
        const _Float16* __restrict__ px = xpad + ((size_t)(bb * 30 + yy) * 30 + xx) * 64;
        const _Float16* __restrict__ pw = wrow + (size_t)tap * 2048 * 64;
        acc = WMMA_F16(load_row16(px, g), load_row16(pw, g), acc);
        acc = WMMA_F16(load_row16(px + 32, g), load_row16(pw + 32, g), acc);
        ++tap;
      }
    }
  }
  const int oc = oc0 + lm;
  const float bias = cb0[oc] + cb1[oc] + cb2[oc] + cb3[oc] + cb4[oc] + cb5[oc];
#pragma unroll
  for (int r = 0; r < 8; ++r) {
    const int m = r + 8 * g;
    const float v = (acc[r] + bias) * (1.0f / 6.0f);
    xb_h[(size_t)(p0 + m) * 2048 + oc] = (_Float16)silu_f(v);
  }
}

// ---------------------------------------------------------------------------
// out[M,N] = A[M,K](f16) @ W[N,K]^T(f16), f32 out.  grid (M/16, N/256), block 256.
// ---------------------------------------------------------------------------
__global__ void gemm_a_wt(const _Float16* __restrict__ A, const _Float16* __restrict__ W,
                          float* __restrict__ out, int M, int N, int K) {
  const int lane = threadIdx.x & 31, wave = threadIdx.x >> 5;
  const int lm = lane & 15, g = lane >> 4;
  const int m0 = blockIdx.x * 16;
  const int nbase = blockIdx.y * 256 + wave * 32;
  const _Float16* __restrict__ rowA = A + (size_t)(m0 + lm) * K;
#pragma unroll
  for (int nt = 0; nt < 2; ++nt) {
    const int n0 = nbase + nt * 16;
    const _Float16* __restrict__ rowW = W + (size_t)(n0 + lm) * K;
    v8f c = {};
    for (int k0 = 0; k0 < K; k0 += 32) {
      __builtin_prefetch(rowA + k0 + 256, 0, 1);
      __builtin_prefetch(rowW + k0 + 256, 0, 1);
      c = WMMA_F16(load_row16(rowA + k0, g), load_row16(rowW + k0, g), c);
    }
#pragma unroll
    for (int r = 0; r < 8; ++r)
      out[(size_t)(m0 + r + 8 * g) * N + n0 + lm] = c[r];
  }
}

// ---------------------------------------------------------------------------
// TTT inner-loop scan.  grid = 16 (b,h) blocks, block 256 (8 waves).
// f16 state in consumer-friendly layouts; LDS transposed shadows make the
// outer-product update fragments contiguous.  TDM stages each chunk.
// ---------------------------------------------------------------------------
#define TTT_SMEM_BYTES (227328)

__global__ void ttt_scan(const float* __restrict__ q, const float* __restrict__ k,
                         const float* __restrict__ v,
                         _Float16* __restrict__ W1cs, _Float16* __restrict__ W2rs,
                         _Float16* __restrict__ W2cs,
                         float* __restrict__ b1s, float* __restrict__ b2s,
                         const float* __restrict__ lnw, const float* __restrict__ lnb,
                         _Float16* __restrict__ o_h) {
  extern __shared__ char smem[];
  _Float16* s_xq   = (_Float16*)smem;              // [16][256]
  _Float16* s_xk   = s_xq + 16 * 256;              // [16][256]
  _Float16* s_xv   = s_xk + 16 * 256;              // [16][256]
  _Float16* s_xkT  = s_xv + 16 * 256;              // [256][16]
  _Float16* s_z1   = s_xkT + 256 * 16;             // [16][1024]
  _Float16* s_a1   = s_z1 + 16 * 1024;             // [16][1024]
  _Float16* s_a1T  = s_a1 + 16 * 1024;             // [1024][16]
  _Float16* s_gz1T = s_a1T + 1024 * 16;            // [1024][16]
  _Float16* s_gz2  = s_gz1T + 1024 * 16;           // [16][256]
  _Float16* s_gz2T = s_gz2 + 16 * 256;             // [256][16]
  float*    s_z2   = (float*)(s_gz2T + 256 * 16);  // [16][256] f32
  float*    s_stg  = s_z2 + 16 * 256;              // [3][10][256] f32 staging

  const int bh = blockIdx.x;
  const int bb = bh >> 3, h = bh & 7;
  _Float16* __restrict__ W1c = W1cs + (size_t)bh * 1024 * 256;
  _Float16* __restrict__ W2r = W2rs + (size_t)bh * 1024 * 256;
  _Float16* __restrict__ W2c = W2cs + (size_t)bh * 256 * 1024;
  float* __restrict__ b1 = b1s + bh * 1024;
  float* __restrict__ b2 = b2s + bh * 256;
  const float* __restrict__ gw = lnw + h * 256;
  const float* __restrict__ gb = lnb + h * 256;

  const int tid = threadIdx.x;
  const int lane = tid & 31, wave = tid >> 5;
  const int lm = lane & 15, g = tid >> 4 & 1;
  const float eta = 0.1f / 10.0f;  // BASE_LR / MB

  for (int c = 0; c < 40; ++c) {
    const int t0 = c * 10;
    const size_t rowoff = (size_t)(bb * 400 + t0) * 2048 + h * 256;
#if HAVE_TDM
    if (wave == 0) {
      const unsigned lds0 = (unsigned)(size_t)(void*)s_stg;
      tdm_load_2d_f32(lds0,          q + rowoff, 256, 10, 2048, 800, 2048);
      tdm_load_2d_f32(lds0 + 10240,  k + rowoff, 256, 10, 2048, 800, 2048);
      tdm_load_2d_f32(lds0 + 20480,  v + rowoff, 256, 10, 2048, 800, 2048);
      __builtin_amdgcn_s_wait_tensorcnt(0);
    }
    __syncthreads();
#endif
    // ---- S0: convert chunk to f16 (+ transposed xk), zero gz2/gz2T ----
    for (int idx = tid; idx < 16 * 256; idx += 256) {
      const int r = idx >> 8, d = idx & 255;
      float qq = 0.f, kk2 = 0.f, vv = 0.f;
      if (r < 10) {
#if HAVE_TDM
        qq = s_stg[r * 256 + d]; kk2 = s_stg[2560 + r * 256 + d]; vv = s_stg[5120 + r * 256 + d];
#else
        qq = q[rowoff + (size_t)r * 2048 + d];
        kk2 = k[rowoff + (size_t)r * 2048 + d];
        vv = v[rowoff + (size_t)r * 2048 + d];
#endif
      }
      s_xq[idx] = (_Float16)qq;
      s_xk[idx] = (_Float16)kk2;
      s_xkT[d * 16 + r] = (_Float16)kk2;
      s_xv[idx] = (_Float16)vv;
      s_gz2[idx] = (_Float16)0.0f;
      s_gz2T[idx] = (_Float16)0.0f;
    }
    __syncthreads();

    // ---- S1: Z1 = xk @ W1 + b1  (16 x 1024, K=256); A1 = gelu(Z1) ----
    for (int nt = wave; nt < 64; nt += 8) {
      const int n0 = nt * 16;
      const _Float16* __restrict__ wrow = W1c + (size_t)(n0 + lm) * 256;
      v8f cc = {};
      for (int k0 = 0; k0 < 256; k0 += 32)
        cc = WMMA_F16(load_row16(s_xk + lm * 256 + k0, g), load_row16(wrow + k0, g), cc);
      const float bias = b1[n0 + lm];
#pragma unroll
      for (int r = 0; r < 8; ++r) {
        const int m = r + 8 * g;
        const float z = cc[r] + bias;
        const float a = gelu_f(z);
        s_z1[m * 1024 + n0 + lm] = (_Float16)z;
        s_a1[m * 1024 + n0 + lm] = (_Float16)a;
        s_a1T[(n0 + lm) * 16 + m] = (_Float16)a;
      }
    }
    __syncthreads();

    // ---- S2: Z2 = A1 @ W2 + b2  (16 x 256, K=1024) ----
    for (int nt = wave; nt < 16; nt += 8) {
      const int n0 = nt * 16;
      const _Float16* __restrict__ wrow = W2c + (size_t)(n0 + lm) * 1024;
      v8f cc = {};
      for (int k0 = 0; k0 < 1024; k0 += 32)
        cc = WMMA_F16(load_row16(s_a1 + lm * 1024 + k0, g), load_row16(wrow + k0, g), cc);
      const float bias = b2[n0 + lm];
#pragma unroll
      for (int r = 0; r < 8; ++r)
        s_z2[(r + 8 * g) * 256 + n0 + lm] = cc[r] + bias;
    }
    __syncthreads();

    // ---- S3: fused LN-l2 backward -> gZ2 (rows 0..9) ----
    for (int r = wave; r < 10; r += 8) {
      float s = 0.f, ss = 0.f;
      for (int d = lane; d < 256; d += 32) { const float z = s_z2[r * 256 + d]; s += z; ss += z * z; }
      for (int o = 16; o; o >>= 1) { s += __shfl_xor(s, o, 32); ss += __shfl_xor(ss, o, 32); }
      const float mu = s * (1.0f / 256.0f);
      const float istd = rsqrtf(ss * (1.0f / 256.0f) - mu * mu + 1e-6f);
      float s1 = 0.f, s2 = 0.f;
      for (int d = lane; d < 256; d += 32) {
        const float z = s_z2[r * 256 + d];
        const float xh = (z - mu) * istd;
        const float tgt = (float)s_xv[r * 256 + d] - (float)s_xk[r * 256 + d];
        const float gxh = ((gw[d] * xh + gb[d]) - tgt) * gw[d];
        s1 += gxh; s2 += gxh * xh;
      }
      for (int o = 16; o; o >>= 1) { s1 += __shfl_xor(s1, o, 32); s2 += __shfl_xor(s2, o, 32); }
      s1 *= (1.0f / 256.0f); s2 *= (1.0f / 256.0f);
      for (int d = lane; d < 256; d += 32) {
        const float z = s_z2[r * 256 + d];
        const float xh = (z - mu) * istd;
        const float gxh = ((gw[d] * xh + gb[d]) - ((float)s_xv[r * 256 + d] - (float)s_xk[r * 256 + d])) * gw[d];
        const float gz2 = (gxh - s1 - xh * s2) * istd;
        s_gz2[r * 256 + d] = (_Float16)gz2;
        s_gz2T[d * 16 + r] = (_Float16)gz2;
      }
    }
    __syncthreads();

    // ---- S4: gZ1 = (gZ2 @ W2^T) * gelu'(Z1)  (16 x 1024, K=256) ----
    for (int nt = wave; nt < 64; nt += 8) {
      const int n0 = nt * 16;
      const _Float16* __restrict__ wrow = W2r + (size_t)(n0 + lm) * 256;
      v8f cc = {};
      for (int k0 = 0; k0 < 256; k0 += 32)
        cc = WMMA_F16(load_row16(s_gz2 + lm * 256 + k0, g), load_row16(wrow + k0, g), cc);
#pragma unroll
      for (int r = 0; r < 8; ++r) {
        const int m = r + 8 * g;
        const float z = (float)s_z1[m * 1024 + n0 + lm];
        s_gz1T[(n0 + lm) * 16 + m] = (_Float16)(cc[r] * gelu_bwd_f(z));
      }
    }
    __syncthreads();

    // ---- S5a: W1 -= eta * xk^T @ gZ1  (256 x 1024, K=16 pad 32) ----
    for (int tIdx = wave; tIdx < 16 * 64; tIdx += 8) {
      const int m0 = (tIdx >> 6) * 16, n0 = (tIdx & 63) * 16;
      v8f cc = {};
      cc = WMMA_F16(load_t16(s_xkT + (m0 + lm) * 16, g),
                    load_t16(s_gz1T + (n0 + lm) * 16, g), cc);
      _Float16* __restrict__ wcol = W1c + (size_t)(n0 + lm) * 256 + m0 + 8 * g;
#pragma unroll
      for (int r = 0; r < 8; ++r)
        wcol[r] = (_Float16)((float)wcol[r] - eta * cc[r]);
    }
    // ---- S5b: W2 -= eta * A1^T @ gZ2  (1024 x 256), update both layouts ----
    for (int tIdx = wave; tIdx < 64 * 16; tIdx += 8) {
      const int m0 = (tIdx >> 4) * 16, n0 = (tIdx & 15) * 16;
      v8f cc = {};
      cc = WMMA_F16(load_t16(s_a1T + (m0 + lm) * 16, g),
                    load_t16(s_gz2T + (n0 + lm) * 16, g), cc);
      _Float16* __restrict__ wcol = W2c + (size_t)(n0 + lm) * 1024 + m0 + 8 * g;
#pragma unroll
      for (int r = 0; r < 8; ++r) {
        const float d = eta * cc[r];
        const int m = m0 + r + 8 * g;
        const size_t ridx = (size_t)m * 256 + n0 + lm;
        W2r[ridx] = (_Float16)((float)W2r[ridx] - d);
        wcol[r] = (_Float16)((float)wcol[r] - d);
      }
    }
    // ---- S5c: bias updates (contiguous via transposed shadows) ----
    for (int col = tid; col < 1024; col += 256) {
      float s = 0.f;
#pragma unroll
      for (int r = 0; r < 10; ++r) s += (float)s_gz1T[col * 16 + r];
      b1[col] -= eta * s;
    }
    {
      float s = 0.f;
#pragma unroll
      for (int r = 0; r < 10; ++r) s += (float)s_gz2T[tid * 16 + r];
      b2[tid] -= eta * s;
    }
    __threadfence();
    __syncthreads();

    // ---- S6a: A1q = gelu(xq @ W1n + b1n) ----
    for (int nt = wave; nt < 64; nt += 8) {
      const int n0 = nt * 16;
      const _Float16* __restrict__ wrow = W1c + (size_t)(n0 + lm) * 256;
      v8f cc = {};
      for (int k0 = 0; k0 < 256; k0 += 32)
        cc = WMMA_F16(load_row16(s_xq + lm * 256 + k0, g), load_row16(wrow + k0, g), cc);
      const float bias = b1[n0 + lm];
#pragma unroll
      for (int r = 0; r < 8; ++r)
        s_a1[(r + 8 * g) * 1024 + n0 + lm] = (_Float16)gelu_f(cc[r] + bias);
    }
    __syncthreads();

    // ---- S6b: Z2q = A1q @ W2n + b2n ----
    for (int nt = wave; nt < 16; nt += 8) {
      const int n0 = nt * 16;
      const _Float16* __restrict__ wrow = W2c + (size_t)(n0 + lm) * 1024;
      v8f cc = {};
      for (int k0 = 0; k0 < 1024; k0 += 32)
        cc = WMMA_F16(load_row16(s_a1 + lm * 1024 + k0, g), load_row16(wrow + k0, g), cc);
      const float bias = b2[n0 + lm];
#pragma unroll
      for (int r = 0; r < 8; ++r)
        s_z2[(r + 8 * g) * 256 + n0 + lm] = cc[r] + bias;
    }
    __syncthreads();

    // ---- S6c: post-LN + residual -> o (f16) ----
    for (int r = wave; r < 10; r += 8) {
      float s = 0.f, ss = 0.f;
      for (int d = lane; d < 256; d += 32) { const float z = s_z2[r * 256 + d]; s += z; ss += z * z; }
      for (int o = 16; o; o >>= 1) { s += __shfl_xor(s, o, 32); ss += __shfl_xor(ss, o, 32); }
      const float mu = s * (1.0f / 256.0f);
      const float istd = rsqrtf(ss * (1.0f / 256.0f) - mu * mu + 1e-6f);
      for (int d = lane; d < 256; d += 32) {
        const float xh = (s_z2[r * 256 + d] - mu) * istd;
        const float outv = (float)s_xq[r * 256 + d] + gw[d] * xh + gb[d];
        o_h[rowoff + (size_t)r * 2048 + d] = (_Float16)outv;
      }
    }
    __syncthreads();
  }
}

// ---------------------------------------------------------------------------
extern "C" void kernel_launch(void* const* d_in, const int* in_sizes, int n_in,
                              void* d_out, int out_size, void* d_ws, size_t ws_size,
                              hipStream_t stream) {
  const float* x = (const float*)d_in[0];
  const float* cw[6]; const float* cb[6];
  for (int i = 0; i < 6; ++i) { cw[i] = (const float*)d_in[2 + 2 * i]; cb[i] = (const float*)d_in[3 + 2 * i]; }
  const float* wq = (const float*)d_in[14];
  const float* wk = (const float*)d_in[15];
  const float* wv = (const float*)d_in[16];
  const float* wo = (const float*)d_in[17];
  const float* W1 = (const float*)d_in[18];
  const float* b1 = (const float*)d_in[19];
  const float* W2 = (const float*)d_in[20];
  const float* b2 = (const float*)d_in[21];
  const float* lnw = (const float*)d_in[22];
  const float* lnb = (const float*)d_in[23];

  char* ws = (char*)d_ws;
  size_t off = 0;
  auto alloc = [&](size_t bytes) -> void* {
    void* p = ws + off;
    off += (bytes + 255) & ~(size_t)255;
    return p;
  };
  const int MT = 800, D = 2048;
  _Float16* xpad = (_Float16*)alloc((size_t)2 * 30 * 30 * 64 * 2);
  _Float16* wt   = (_Float16*)alloc((size_t)286 * 2048 * 64 * 2);
  _Float16* xb_h = (_Float16*)alloc((size_t)MT * D * 2);
  float* qf = (float*)alloc((size_t)MT * D * 4);
  float* kf = (float*)alloc((size_t)MT * D * 4);
  float* vf = (float*)alloc((size_t)MT * D * 4);
  _Float16* oh  = (_Float16*)alloc((size_t)MT * D * 2);
  _Float16* wqh = (_Float16*)alloc((size_t)D * D * 2);
  _Float16* wkh = (_Float16*)alloc((size_t)D * D * 2);
  _Float16* wvh = (_Float16*)alloc((size_t)D * D * 2);
  _Float16* woh = (_Float16*)alloc((size_t)D * D * 2);
  _Float16* W1c = (_Float16*)alloc((size_t)16 * 1024 * 256 * 2);
  _Float16* W2r = (_Float16*)alloc((size_t)16 * 1024 * 256 * 2);
  _Float16* W2c = (_Float16*)alloc((size_t)16 * 256 * 1024 * 2);
  float* b1s = (float*)alloc((size_t)16 * 1024 * 4);
  float* b2s = (float*)alloc((size_t)16 * 256 * 4);

  pad_input<<<(2 * 30 * 30 * 64 + 255) / 256, 256, 0, stream>>>(x, xpad);
  {
    int tapbase = 0;
    for (int kern = 0; kern < 6; ++kern) {
      const int kk = 2 * kern + 1, k2 = kk * kk;
      const size_t n = (size_t)2048 * 64 * k2;
      reorg_convw<<<(unsigned)((n + 255) / 256), 256, 0, stream>>>(
          cw[kern], wt + (size_t)tapbase * 2048 * 64, k2);
      tapbase += k2;
    }
  }
  const int nW = D * D;
  cvt_f32_to_f16<<<(nW + 255) / 256, 256, 0, stream>>>(wq, wqh, nW);
  cvt_f32_to_f16<<<(nW + 255) / 256, 256, 0, stream>>>(wk, wkh, nW);
  cvt_f32_to_f16<<<(nW + 255) / 256, 256, 0, stream>>>(wv, wvh, nW);
  cvt_f32_to_f16<<<(nW + 255) / 256, 256, 0, stream>>>(wo, woh, nW);
  init_ttt_state<<<16384, 256, 0, stream>>>(W1, b1, W2, b2, W1c, W2r, W2c, b1s, b2s);

  conv_ensemble_silu<<<dim3(50, 32), 128, 0, stream>>>(
      xpad, wt, cb[0], cb[1], cb[2], cb[3], cb[4], cb[5], xb_h);

  gemm_a_wt<<<dim3(50, 8), 256, 0, stream>>>(xb_h, wqh, qf, MT, D, D);
  gemm_a_wt<<<dim3(50, 8), 256, 0, stream>>>(xb_h, wkh, kf, MT, D, D);
  gemm_a_wt<<<dim3(50, 8), 256, 0, stream>>>(xb_h, wvh, vf, MT, D, D);

  ttt_scan<<<16, 256, TTT_SMEM_BYTES, stream>>>(qf, kf, vf, W1c, W2r, W2c, b1s, b2s, lnw, lnb, oh);

  gemm_a_wt<<<dim3(50, 8), 256, 0, stream>>>(oh, woh, (float*)d_out, MT, D, D);
}
